// DiscretePixelCNN_16363825398098
// MI455X (gfx1250) — matrix operand order repaired and checked by reading, not compiled, not run on hardware
//
#include <hip/hip_runtime.h>
#include <hip/hip_bf16.h>

typedef __attribute__((ext_vector_type(16))) _Float16 v16h;
typedef __attribute__((ext_vector_type(8)))  _Float16 v8h;
typedef __attribute__((ext_vector_type(8)))  float    v8f;

#define PB 32
#define PH 64
#define PW 64
#define HID 64
#define PHP 67   // padded height: 3-row halo on top (dy in [-3,0])
#define PWP 70   // padded width: 3-col halo left and right (dx in [-3,3])

static constexpr size_t ACTP_HALFS = (size_t)PB * PHP * PWP * HID;  // 9,605,120
static constexpr size_t ACTP_BYTES = ACTP_HALFS * 2;                // 19,210,240
static constexpr size_t W16_HALFS  = (size_t)5 * HID * 49 * HID;    // 1,003,520
static constexpr size_t W16_BYTES  = W16_HALFS * 2;                 // 2,007,040

// ---------------- zero both padded activation buffers (halo stays zero all launch)
__global__ __launch_bounds__(256) void zero_buf(uint4* __restrict__ p, unsigned n16)
{
  unsigned i = blockIdx.x * 256 + threadIdx.x;
  if (i < n16) {
    uint4 z; z.x = 0; z.y = 0; z.z = 0; z.w = 0;
    p[i] = z;
  }
}

// ---------------- prep: mask-B f16 weights for layers 1..5, layout [l][oc][tap49][ic]
__global__ __launch_bounds__(256) void prep_w16(
    const float* __restrict__ w1, const float* __restrict__ w2,
    const float* __restrict__ w3, const float* __restrict__ w4,
    const float* __restrict__ w5, _Float16* __restrict__ w16)
{
  size_t idx = (size_t)blockIdx.x * 256 + threadIdx.x;
  if (idx >= W16_HALFS) return;
  int ic  = (int)(idx & 63);
  int tap = (int)((idx >> 6) % 49);
  int oc  = (int)(((idx >> 6) / 49) & 63);
  int l   = (int)(idx / ((size_t)64 * 49 * 64));
  const float* w = (l == 0) ? w1 : (l == 1) ? w2 : (l == 2) ? w3 : (l == 3) ? w4 : w5;
  int row = tap / 7, col = tap % 7;
  bool keep = (row < 3) || (row == 3 && col <= 3);   // mask B (center included)
  float v = keep ? w[((size_t)oc * HID + ic) * 49 + tap] : 0.f;
  w16[idx] = (_Float16)v;
}

// ---------------- prep: mask-A layer-0 weights compacted to [oc][24] f32
__global__ __launch_bounds__(256) void prep_w0(
    const float* __restrict__ w0, float* __restrict__ w0m)
{
  int idx = blockIdx.x * 256 + threadIdx.x;
  if (idx >= 64 * 24) return;
  int t = idx % 24, oc = idx / 24;
  int dy = (t < 21) ? (t / 7 - 3) : 0;
  int dx = (t < 21) ? (t % 7 - 3) : (t - 21) - 3;   // row 3: dx = -3..-1 (no center)
  int tap = (dy + 3) * 7 + (dx + 3);
  w0m[idx] = w0[oc * 49 + tap];
}

// ---------------- layer 0: 1->64 channels, mask A, VALU (tiny FLOP count)
__global__ __launch_bounds__(256) void layer0(
    const float* __restrict__ sample, const float* __restrict__ w0m,
    const float* __restrict__ bias, _Float16* __restrict__ out)
{
  __shared__ float lw[64 * 24];
  __shared__ float lb[64];
  for (int i = threadIdx.x; i < 64 * 24; i += 256) lw[i] = w0m[i];
  if (threadIdx.x < 64) lb[threadIdx.x] = bias[threadIdx.x];
  __syncthreads();

  int idx = blockIdx.x * 256 + threadIdx.x;          // one (b,y,x) per thread
  int b = idx / (PH * PW);
  int rem = idx % (PH * PW);
  int y = rem / PW, x = rem % PW;

  float s[24];
#pragma unroll
  for (int t = 0; t < 24; ++t) {
    int dy = (t < 21) ? (t / 7 - 3) : 0;
    int dx = (t < 21) ? (t % 7 - 3) : (t - 21) - 3;
    int yy = y + dy, xx = x + dx;
    s[t] = (yy >= 0 && xx >= 0 && xx < PW) ? sample[((size_t)b * PH + yy) * PW + xx] : 0.f;
  }

  _Float16* op = out + (((size_t)b * PHP + y + 3) * PWP + x + 3) * HID;
#pragma unroll
  for (int og = 0; og < 8; ++og) {
    v8h hv;
#pragma unroll
    for (int r = 0; r < 8; ++r) {
      int oc = og * 8 + r;
      float acc = lb[oc];
#pragma unroll
      for (int t = 0; t < 24; ++t) acc += lw[oc * 24 + t] * s[t];
      hv[r] = (_Float16)fmaxf(acc, 0.f);
    }
    *(v8h*)(op + og * 8) = hv;
  }
}

// ---------------- layers 1..5: 64->64 mask-B conv as implicit GEMM via WMMA
// block = 128 threads = 4 wave32s; wave w computes oc tile [16w,16w+16) x 32 positions.
// Padded input => every fragment load is unconditional and in-bounds.
__global__ __launch_bounds__(128) void conv_mask_b(
    const _Float16* __restrict__ in, _Float16* __restrict__ out,
    const _Float16* __restrict__ w, const float* __restrict__ bias)
{
  const int lane = threadIdx.x & 31;
  const int wave = threadIdx.x >> 5;
  const int n    = lane & 15;     // column / N index
  const int hi   = lane >> 4;     // lane-half selector
  const int x0   = blockIdx.x * 32;   // two 16-wide N tiles
  const int y    = blockIdx.y;
  const int b    = blockIdx.z;
  const int m    = wave * 16 + n; // out-channel row of A

  v8f acc0 = {}, acc1 = {};
  const _Float16* aBase = w + (size_t)m * 49 * HID;
  // center position of this lane's column in the padded plane, K-offset hi*16
  const _Float16* bBase =
      in + (((size_t)b * PHP + (y + 3)) * PWP + 3 + x0 + n) * HID + hi * 16;

#pragma unroll
  for (int t = 0; t < 25; ++t) {
    const int dy  = (t < 21) ? (t / 7 - 3) : 0;
    const int dx  = (t < 21) ? (t % 7 - 3) : (t - 21) - 3;   // row 3: dx = -3..0
    const int tap = (dy + 3) * 7 + (dx + 3);
    const _Float16* rp = bBase + (ptrdiff_t)(dy * PWP + dx) * HID;  // signed offset!
    const v8h* q = (const v8h*)(aBase + (size_t)tap * HID);
#pragma unroll
    for (int c = 0; c < 2; ++c) {          // two K=32 ic chunks
      // A fragment: lane<16 -> K {0..15}+32c ; lane>=16 -> K {16..31}+32c
      v8h al = q[4 * c + 2 * hi], ah = q[4 * c + 2 * hi + 1];
      v16h a;
#pragma unroll
      for (int i = 0; i < 8; ++i) { a[i] = al[i]; a[i + 8] = ah[i]; }
      // B fragments for both N tiles: lane holds 16 contiguous K halfs
      const v8h* p0 = (const v8h*)(rp + c * 32);
      const v8h* p1 = (const v8h*)(rp + 16 * HID + c * 32);
      v8h b0l = p0[0], b0h = p0[1], b1l = p1[0], b1h = p1[1];
      v16h b0, b1;
#pragma unroll
      for (int i = 0; i < 8; ++i) {
        b0[i] = b0l[i]; b0[i + 8] = b0h[i];
        b1[i] = b1l[i]; b1[i + 8] = b1h[i];
      }
      acc0 = __builtin_amdgcn_wmma_f32_16x16x32_f16(false, a, false, b0, (short)0, acc0, false, false);
      acc1 = __builtin_amdgcn_wmma_f32_16x16x32_f16(false, a, false, b1, (short)0, acc1, false, false);
    }
  }

  // D layout: VGPR r -> row m = hi*8 + r, col n; fuse bias + ReLU, pack f16, b128 stores
  const int ocb = wave * 16 + hi * 8;
  const float* bp = bias + ocb;
  v8h hv0, hv1;
#pragma unroll
  for (int r = 0; r < 8; ++r) {
    float bb = bp[r];
    hv0[r] = (_Float16)fmaxf(acc0[r] + bb, 0.f);
    hv1[r] = (_Float16)fmaxf(acc1[r] + bb, 0.f);
  }
  _Float16* op = out + (((size_t)b * PHP + (y + 3)) * PWP + 3 + x0 + n) * HID + ocb;
  *(v8h*)op = hv0;
  *(v8h*)(op + 16 * HID) = hv1;
}

// ---------------- head: 1x1 conv (64->2), log-softmax over K=2, gather, per-batch sum
__global__ __launch_bounds__(256) void head(
    const _Float16* __restrict__ h, const float* __restrict__ sample,
    const float* __restrict__ w_out, const float* __restrict__ b_out,
    float* __restrict__ outp)
{
  __shared__ float lw[2 * HID];
  __shared__ float lb2[2];
  __shared__ float red[256];
  const int tid = threadIdx.x;
  const int b = blockIdx.x;
  if (tid < 2 * HID) lw[tid] = w_out[tid];
  if (tid < 2) lb2[tid] = b_out[tid];
  __syncthreads();

  float sum = 0.f;
  for (int p = tid; p < PH * PW; p += 256) {
    int y = p >> 6, x = p & 63;
    const _Float16* hp = h + (((size_t)b * PHP + y + 3) * PWP + x + 3) * HID;
    float d0 = lb2[0], d1 = lb2[1];
#pragma unroll
    for (int c = 0; c < HID; ++c) {
      float hv = (float)hp[c];
      d0 += lw[c] * hv;
      d1 += lw[HID + c] * hv;
    }
    float sv   = sample[(size_t)b * PH * PW + p];
    float lsel = (sv > 0.5f) ? d1 : d0;
    float mx   = fmaxf(d0, d1);
    float lse  = mx + __logf(__expf(d0 - mx) + __expf(d1 - mx));
    sum += lsel - lse;
  }
  red[tid] = sum;
  __syncthreads();
  for (int s = 128; s > 0; s >>= 1) {
    if (tid < s) red[tid] += red[tid + s];
    __syncthreads();
  }
  if (tid == 0) outp[b] = red[0];
}

extern "C" void kernel_launch(void* const* d_in, const int* in_sizes, int n_in,
                              void* d_out, int out_size, void* d_ws, size_t ws_size,
                              hipStream_t stream)
{
  // inputs: [0]=sample, [1..6]=ws[0..5], [7..12]=bs[0..5], [13]=w_out, [14]=b_out
  const float* sample = (const float*)d_in[0];
  const float* wsL[6];
  const float* bsL[6];
  for (int i = 0; i < 6; ++i) wsL[i] = (const float*)d_in[1 + i];
  for (int i = 0; i < 6; ++i) bsL[i] = (const float*)d_in[7 + i];
  const float* w_out = (const float*)d_in[13];
  const float* b_out = (const float*)d_in[14];

  char* base = (char*)d_ws;
  _Float16* act0 = (_Float16*)base;
  _Float16* act1 = (_Float16*)(base + ACTP_BYTES);
  _Float16* w16  = (_Float16*)(base + 2 * ACTP_BYTES);
  float*    w0m  = (float*)(base + 2 * ACTP_BYTES + W16_BYTES);

  // zero both padded activation buffers (provides the SAME-padding halo)
  const unsigned n16 = (unsigned)((2 * ACTP_BYTES) / 16);
  zero_buf<<<(n16 + 255) / 256, 256, 0, stream>>>((uint4*)base, n16);

  prep_w16<<<(unsigned)((W16_HALFS + 255) / 256), 256, 0, stream>>>(
      wsL[1], wsL[2], wsL[3], wsL[4], wsL[5], w16);
  prep_w0<<<6, 256, 0, stream>>>(wsL[0], w0m);

  layer0<<<(PB * PH * PW) / 256, 256, 0, stream>>>(sample, w0m, bsL[0], act0);

  for (int l = 1; l < 6; ++l) {
    const _Float16* inp  = (l & 1) ? act0 : act1;
    _Float16*       outp = (l & 1) ? act1 : act0;
    conv_mask_b<<<dim3(2, PH, PB), 128, 0, stream>>>(
        inp, outp, w16 + (size_t)(l - 1) * HID * 49 * HID, bsL[l]);
  }

  head<<<PB, 256, 0, stream>>>(act1, sample, w_out, b_out, (float*)d_out);

  (void)in_sizes; (void)n_in; (void)out_size; (void)ws_size;
}
